// HybridRegressor_65481071404974
// MI455X (gfx1250) — matrix operand order repaired and checked by reading, not compiled, not run on hardware
//
#include <hip/hip_runtime.h>

typedef __attribute__((ext_vector_type(16))) _Float16 v16h;
typedef __attribute__((ext_vector_type(8)))  float    v8f;
typedef __attribute__((ext_vector_type(4)))  float    f32x4;

#define N_RAND 30
#define FEAT 256

// branchless tanh: 1 - 2/(exp(2|x|)+1), sign restored (v_exp_f32 path, no EXEC divergence)
__device__ __forceinline__ float fast_tanh(float v) {
  const float ax = fabsf(v);
  const float e  = __expf(2.0f * ax);
  const float r  = 1.0f - 2.0f / (e + 1.0f);
  return copysignf(r, v);
}

__global__ __launch_bounds__(256)
void hybrid_qreg_kernel(const float* __restrict__ x,        // [B, 256]
                        const float* __restrict__ enc_w,    // [8, 256]
                        const float* __restrict__ enc_b,    // [8]
                        const float* __restrict__ rp,       // [30]
                        const float* __restrict__ head_w,   // [1, 4]
                        const float* __restrict__ head_b,   // [1]
                        float* __restrict__ out)            // [B]
{
  __shared__ float  params_lds[8][32 * 17];  // per-wave padded transpose buffer
  __shared__ float2 U_lds[16][16];           // U[i][t]: amplitude i of random-layer column t
  __shared__ float2 Moff[16][16];            // j<k: 2 * M_jk  (M = U^H diag(g) U)
  __shared__ float  Mdiag[16];               // real diagonal of M

  const int tid  = threadIdx.x;
  const int wave = tid >> 5;
  const int lane = tid & 31;
  const int h    = lane >> 4;     // half-wave group (K-chunk selector)
  const int m    = lane & 15;     // A-row within tile / B&C column
  const size_t rowBase = (size_t)blockIdx.x * 256 + (size_t)wave * 32;

  // -------- Phase 1: encoder GEMM via v_wmma_f32_16x16x32_f16 --------
  v8f acc0 = {};  // rows rowBase + 0..15
  v8f acc1 = {};  // rows rowBase + 16..31
  const float* xr0 = x + (rowBase + m) * FEAT;
  const float* xr1 = x + (rowBase + 16 + m) * FEAT;
  const float bmask = (m < 8) ? 1.0f : 0.0f;           // zero-pad cols 8..15 (no divergence)
  const float* wrow = enc_w + (m & 7) * FEAT;

  for (int kb = 0; kb < FEAT / 32; ++kb) {
    const int k0 = kb * 32 + h * 8;                    // A: K = {k0..k0+7, k0+16..k0+23}
    f32x4 a00 = *(const f32x4*)(xr0 + k0);
    f32x4 a01 = *(const f32x4*)(xr0 + k0 + 4);
    f32x4 a02 = *(const f32x4*)(xr0 + k0 + 16);
    f32x4 a03 = *(const f32x4*)(xr0 + k0 + 20);
    f32x4 a10 = *(const f32x4*)(xr1 + k0);
    f32x4 a11 = *(const f32x4*)(xr1 + k0 + 4);
    f32x4 a12 = *(const f32x4*)(xr1 + k0 + 16);
    f32x4 a13 = *(const f32x4*)(xr1 + k0 + 20);
    const float* wp = wrow + kb * 32 + h * 8;          // B: col m, K = {wp..wp+7, wp+16..wp+23}
    f32x4 b0 = *(const f32x4*)(wp);
    f32x4 b1 = *(const f32x4*)(wp + 4);
    f32x4 b2 = *(const f32x4*)(wp + 16);
    f32x4 b3 = *(const f32x4*)(wp + 20);

    v16h A0, A1, Bm;
#pragma unroll
    for (int j = 0; j < 4; ++j) {
      A0[j]      = (_Float16)a00[j];  A0[4 + j]  = (_Float16)a01[j];
      A0[8 + j]  = (_Float16)a02[j];  A0[12 + j] = (_Float16)a03[j];
      A1[j]      = (_Float16)a10[j];  A1[4 + j]  = (_Float16)a11[j];
      A1[8 + j]  = (_Float16)a12[j];  A1[12 + j] = (_Float16)a13[j];
      Bm[j]      = (_Float16)(b0[j] * bmask);
      Bm[4 + j]  = (_Float16)(b1[j] * bmask);
      Bm[8 + j]  = (_Float16)(b2[j] * bmask);
      Bm[12 + j] = (_Float16)(b3[j] * bmask);
    }

    acc0 = __builtin_amdgcn_wmma_f32_16x16x32_f16(false, A0, false, Bm, (short)0, acc0, false, false);
    acc1 = __builtin_amdgcn_wmma_f32_16x16x32_f16(false, A1, false, Bm, (short)0, acc1, false, false);
  }

  // C/D layout: lane L, elem r -> (M = r + 8h, N = L%16). Transpose through LDS.
  {
    float* P = &params_lds[wave][0];
#pragma unroll
    for (int r = 0; r < 8; ++r) {
      P[(r + 8 * h) * 17 + m]      = acc0[r];
      P[(16 + r + 8 * h) * 17 + m] = acc1[r];
    }
  }

  // -------- Phase 1b: build U_rand columns (threads 0..15, rolled, once per block) ----
  if (tid < 16) {
#pragma unroll 1
    for (int i = 0; i < 16; ++i) U_lds[i][tid] = make_float2(i == tid ? 1.0f : 0.0f, 0.0f);
#pragma unroll 1
    for (int k = 0; k < N_RAND; ++k) {
      const int st = 8 >> (k & 3);      // wire k%4 <-> bit 3-(k%4)
      const int g  = k % 3;             // RX / RY / RZ
      const float th = rp[k] * 0.5f;
      const float c = cosf(th), s = sinf(th);
#pragma unroll 1
      for (int i = 0; i < 16; ++i) {
        if (i & st) continue;
        const int j = i + st;
        const float2 a0 = U_lds[i][tid], a1 = U_lds[j][tid];
        float2 n0, n1;
        if (g == 0) {        // RX: [[c,-is],[-is,c]]
          n0.x = c * a0.x + s * a1.y;  n0.y = c * a0.y - s * a1.x;
          n1.x = c * a1.x + s * a0.y;  n1.y = c * a1.y - s * a0.x;
        } else if (g == 1) { // RY: [[c,-s],[s,c]]
          n0.x = c * a0.x - s * a1.x;  n0.y = c * a0.y - s * a1.y;
          n1.x = s * a0.x + c * a1.x;  n1.y = s * a0.y + c * a1.y;
        } else {             // RZ: diag(e^-i th, e^+i th)
          n0.x = c * a0.x + s * a0.y;  n0.y = c * a0.y - s * a0.x;
          n1.x = c * a1.x - s * a1.y;  n1.y = c * a1.y + s * a1.x;
        }
        U_lds[i][tid] = n0;  U_lds[j][tid] = n1;
      }
    }
  }
  __syncthreads();

  // -------- Phase 1c: M = U^H diag(g) U (one (j,k) entry per thread) --------
  {
    const int j = tid >> 4, k = tid & 15;
    if (j <= k) {
      const float hw0 = head_w[0], hw1 = head_w[1], hw2 = head_w[2], hw3 = head_w[3];
      float mre = 0.0f, mim = 0.0f;
#pragma unroll 1
      for (int i = 0; i < 16; ++i) {
        const float gi = ((i & 8) ? -hw0 : hw0) + ((i & 4) ? -hw1 : hw1)
                       + ((i & 2) ? -hw2 : hw2) + ((i & 1) ? -hw3 : hw3);
        const float2 uj = U_lds[i][j], uk = U_lds[i][k];
        mre = fmaf(gi, uj.x * uk.x + uj.y * uk.y, mre);
        mim = fmaf(gi, uj.x * uk.y - uj.y * uk.x, mim);
      }
      if (j == k) Mdiag[j] = mre;                          // Hermitian: real diagonal
      else        Moff[j][k] = make_float2(2.0f * mre, 2.0f * mim);  // pre-doubled
    }
  }
  __syncthreads();

  // -------- Phase 2: per-lane encoded product state --------
  float pr[8];
  {
    const float* P = &params_lds[wave][lane * 17];
#pragma unroll
    for (int j = 0; j < 8; ++j) pr[j] = fast_tanh(P[j] + enc_b[j]);
  }

  // psi_w = RY(b)RX(a)|0> = ((cb*ca, sb*sa), (sb*ca, -cb*sa))
  float A0r[4], A0i[4], A1r[4], A1i[4];
#pragma unroll
  for (int w = 0; w < 4; ++w) {
    float sa, ca, sb, cb;
    __sincosf(pr[w] * 0.5f, &sa, &ca);
    __sincosf(pr[w + 4] * 0.5f, &sb, &cb);
    A0r[w] = cb * ca;   A0i[w] = sb * sa;
    A1r[w] = sb * ca;   A1i[w] = -cb * sa;
  }

  // product-state tree: s = (psi0 x psi1) x (psi2 x psi3); wire w <-> bit 3-w
  float s01r[4], s01i[4], s23r[4], s23i[4];
#pragma unroll
  for (int a = 0; a < 2; ++a) {
#pragma unroll
    for (int b = 0; b < 2; ++b) {
      const int idx = a * 2 + b;
      {
        const float xr = a ? A1r[0] : A0r[0], xi = a ? A1i[0] : A0i[0];
        const float yr = b ? A1r[1] : A0r[1], yi = b ? A1i[1] : A0i[1];
        s01r[idx] = xr * yr - xi * yi;  s01i[idx] = xr * yi + xi * yr;
      }
      {
        const float xr = a ? A1r[2] : A0r[2], xi = a ? A1i[2] : A0i[2];
        const float yr = b ? A1r[3] : A0r[3], yi = b ? A1i[3] : A0i[3];
        s23r[idx] = xr * yr - xi * yi;  s23i[idx] = xr * yi + xi * yr;
      }
    }
  }
  float Sr[16], Si[16];
#pragma unroll
  for (int j = 0; j < 16; ++j) {
    const int u = j >> 2, v = j & 3;
    Sr[j] = s01r[u] * s23r[v] - s01i[u] * s23i[v];
    Si[j] = s01r[u] * s23i[v] + s01i[u] * s23r[v];
  }

  // -------- Phase 3: out = s^H M s + head_b (Hermitian quadratic form) --------
  float acc = head_b[0];
#pragma unroll
  for (int j = 0; j < 16; ++j)
    acc = fmaf(Sr[j] * Sr[j] + Si[j] * Si[j], Mdiag[j], acc);
#pragma unroll
  for (int j = 0; j < 16; ++j) {
#pragma unroll
    for (int k = j + 1; k < 16; ++k) {
      const float2 mjk = Moff[j][k];                 // broadcast ds_load_b64
      const float tr = Sr[j] * Sr[k] + Si[j] * Si[k];  // Re(conj(s_j) s_k)
      const float ti = Sr[j] * Si[k] - Si[j] * Sr[k];  // Im(conj(s_j) s_k)
      acc = fmaf(mjk.x, tr, acc);
      acc = fmaf(-mjk.y, ti, acc);
    }
  }

  out[rowBase + lane] = acc;
}

extern "C" void kernel_launch(void* const* d_in, const int* in_sizes, int n_in,
                              void* d_out, int out_size, void* d_ws, size_t ws_size,
                              hipStream_t stream) {
  const float* x      = (const float*)d_in[0];
  const float* enc_w  = (const float*)d_in[1];
  const float* enc_b  = (const float*)d_in[2];
  const float* rp     = (const float*)d_in[3];
  const float* head_w = (const float*)d_in[4];
  const float* head_b = (const float*)d_in[5];
  float* out = (float*)d_out;

  const int B = out_size;              // 262144
  const int blocks = B / 256;          // 256 samples per block (8 waves x 32)
  hybrid_qreg_kernel<<<blocks, 256, 0, stream>>>(x, enc_w, enc_b, rp, head_w, head_b, out);
}
